// crossSpyGR_module_4879082848464
// MI455X (gfx1250) — matrix operand order repaired and checked by reading, compiled, not run on hardware
//
#include <hip/hip_runtime.h>

#define BB   16
#define CC   512
#define HWHW 3136
#define WCH  128
#define CHW  (CC * HWHW)

typedef __bf16 bf16;
typedef __attribute__((ext_vector_type(16))) __bf16 v16bf;
typedef __attribute__((ext_vector_type(8)))  float  v8f;
typedef __attribute__((ext_vector_type(4)))  int    v4i;

#if defined(__HIP_DEVICE_COMPILE__) && __has_builtin(__builtin_amdgcn_global_load_async_to_lds_b128)
#define HAS_ASYNC 1
#else
#define HAS_ASYNC 0
#endif

static __device__ inline v8f wmma_bf16(v16bf a, v16bf b, v8f c) {
  return __builtin_amdgcn_wmma_f32_16x16x32_bf16(false, a, false, b, (short)0, c, false, false);
}

// A fragment: LDS tile A[m][k] (16 x 32, row stride lda). Lane holds row m=lane&15;
// element e -> k = kbase + (e>>3)*16 + (e&7), kbase = 0 (lanes 0-15) or 8 (lanes 16-31).
static __device__ inline v16bf frag_A(const bf16* base, int lda, int lane) {
  const bf16* p = base + (lane & 15) * lda + ((lane < 16) ? 0 : 8);
  v16bf v;
#pragma unroll
  for (int e = 0; e < 16; ++e) v[e] = p[((e >> 3) << 4) + (e & 7)];
  return v;
}

// B fragment: LDS tile stored TRANSPOSED Bt[n][k] (16 x 32, row stride ldb).
// Lane holds col n=lane&15; element e -> k = kbase + (e>>3)*16 + (e&7), kbase = 0 (lanes 0-15) or 8 (lanes 16-31).
static __device__ inline v16bf frag_B(const bf16* base, int ldb, int lane) {
  const bf16* p = base + (lane & 15) * ldb + ((lane < 16) ? 0 : 8);
  v16bf v;
#pragma unroll
  for (int e = 0; e < 16; ++e) v[e] = p[((e >> 3) << 4) + (e & 7)];
  return v;
}

// 16-byte global->LDS copy: async-to-LDS when available (ASYNCcnt path), else manual.
static __device__ inline void copy16_g2l(const bf16* g, bf16* l) {
#if HAS_ASYNC
  __builtin_amdgcn_global_load_async_to_lds_b128(
      (__attribute__((address_space(1))) v4i*)(void*)(const_cast<bf16*>(g)),
      (__attribute__((address_space(3))) v4i*)(void*)l, 0, 0);
#else
  *(uint4*)l = *(const uint4*)g;
#endif
}

static __device__ inline void wait_g2l() {
#if HAS_ASYNC
#if __has_builtin(__builtin_amdgcn_s_wait_asynccnt)
  __builtin_amdgcn_s_wait_asynccnt(0);
#else
  asm volatile("s_wait_asynccnt 0x0" ::: "memory");
#endif
#endif
}

// Copy ROWS x 32 bf16 tile (global row stride gs, LDS row stride ls; both rows 16B-aligned)
template <int ROWS>
static __device__ inline void tile_copy32(const bf16* g, int gs, bf16* l, int ls, int tid) {
#pragma unroll
  for (int e = tid; e < ROWS * 4; e += 256) {
    int r = e >> 2, c8 = (e & 3) << 3;
    copy16_g2l(g + (size_t)r * gs + c8, l + r * ls + c8);
  }
}

// ---------- K0: convert weights to bf16 ----------
__global__ void k_cvt_w(const float* __restrict__ pw, const float* __restrict__ zw,
                        bf16* __restrict__ pwb, bf16* __restrict__ zwb) {
  int i = blockIdx.x * 256 + threadIdx.x;
  if (i < WCH * CC) pwb[i] = (bf16)pw[i];
  if (i < CC * CC)  zwb[i] = (bf16)zw[i];
}

// ---------- K1: pooled[b][c] = mean over HW of y ----------
__global__ __launch_bounds__(256)
void k_pool(const float* __restrict__ y, float* __restrict__ pooled) {
  int b = blockIdx.y, c = blockIdx.x;
  const float* p = y + (size_t)b * CHW + (size_t)c * HWHW;
  float s = 0.f;
  for (int i = threadIdx.x; i < HWHW; i += 256) s += p[i];
  __shared__ float red[256];
  red[threadIdx.x] = s; __syncthreads();
  for (int off = 128; off > 0; off >>= 1) {
    if (threadIdx.x < off) red[threadIdx.x] += red[threadIdx.x + off];
    __syncthreads();
  }
  if (threadIdx.x == 0) pooled[b * CC + c] = red[0] * (1.0f / (float)HWHW);
}

// ---------- K2: rou[b][k] = sigmoid(pooled @ rou_w^T + rou_b) ----------
__global__ void k_rou(const float* __restrict__ pooled, const float* __restrict__ rw,
                      const float* __restrict__ rb, float* __restrict__ rou) {
  int b = blockIdx.x, k = threadIdx.x;   // 128 threads
  const float* pb = pooled + b * CC;
  const float* wr = rw + k * CC;
  float acc = rb[k];
  for (int c = 0; c < CC; ++c) acc += pb[c] * wr[c];
  rou[b * WCH + k] = 1.f / (1.f + expf(-acc));
}

// ---------- K3: phiT[b][128][HW] = relu(phi_w @ X_c + phi_b), bf16 out ----------
__global__ __launch_bounds__(256)
void k_phi_gemm(const float* __restrict__ x, const bf16* __restrict__ phiw,
                const float* __restrict__ phib, bf16* __restrict__ phiT) {
  const int b = blockIdx.y;
  const int n0 = blockIdx.x * 64;
  const int tid = threadIdx.x, lane = tid & 31, w = tid >> 5;
  const int wm = w >> 1, wn = w & 1;         // 4x2 waves, 32x32 subtile each
  __shared__ __align__(16) bf16 sA[128 * 32];
  __shared__ __align__(16) bf16 sB[64 * 40];
  const float* xb = x + (size_t)b * CHW;
  v8f acc[2][2] = {};
  for (int k0 = 0; k0 < CC; k0 += 32) {
    __syncthreads();
    tile_copy32<128>(phiw + k0, CC, sA, 32, tid);   // A: phi_w tile 128x32 (async path)
#pragma unroll
    for (int i = 0; i < 8; ++i) {            // B: X_c tile, store transposed [n][k]
      int e = tid + i * 256, k = e >> 6, n = e & 63;
      sB[n * 40 + k] = (bf16)xb[(size_t)(k0 + k) * HWHW + n0 + n];
    }
    wait_g2l();
    __syncthreads();
#pragma unroll
    for (int sm = 0; sm < 2; ++sm) {
      v16bf a = frag_A(sA + (wm * 32 + sm * 16) * 32, 32, lane);
#pragma unroll
      for (int sn = 0; sn < 2; ++sn) {
        v16bf bb = frag_B(sB + (wn * 32 + sn * 16) * 40, 40, lane);
        acc[sm][sn] = wmma_bf16(a, bb, acc[sm][sn]);
      }
    }
  }
  bf16* pT = phiT + (size_t)b * WCH * HWHW;
#pragma unroll
  for (int sm = 0; sm < 2; ++sm)
#pragma unroll
    for (int sn = 0; sn < 2; ++sn) {
      int nc = n0 + wn * 32 + sn * 16 + (lane & 15);
#pragma unroll
      for (int i = 0; i < 8; ++i) {
        int m = wm * 32 + sm * 16 + i + ((lane < 16) ? 0 : 8);
        float v = fmaxf(acc[sm][sn][i] + phib[m], 0.f);
        pT[(size_t)m * HWHW + nc] = (bf16)v;
      }
    }
}

// ---------- K4: s[b][k] = sum_n phiT; wk = s * rou ----------
__global__ __launch_bounds__(256)
void k_s(const bf16* __restrict__ phiT, const float* __restrict__ rou,
         float* __restrict__ sOut, float* __restrict__ wk) {
  int b = blockIdx.y, k = blockIdx.x;
  const bf16* p = phiT + ((size_t)b * WCH + k) * HWHW;
  float s = 0.f;
  for (int i = threadIdx.x; i < HWHW; i += 256) s += (float)p[i];
  __shared__ float red[256];
  red[threadIdx.x] = s; __syncthreads();
  for (int off = 128; off > 0; off >>= 1) {
    if (threadIdx.x < off) red[threadIdx.x] += red[threadIdx.x + off];
    __syncthreads();
  }
  if (threadIdx.x == 0) {
    sOut[b * WCH + k] = red[0];
    wk[b * WCH + k] = red[0] * rou[b * WCH + k];
  }
}

// ---------- K5: invD[b][n] = rsqrt(sum_k phiT[k][n]*wk[k] + 1e-8) ----------
__global__ __launch_bounds__(256)
void k_invd(const bf16* __restrict__ phiT, const float* __restrict__ wk,
            float* __restrict__ invD) {
  int b = blockIdx.y;
  int n = blockIdx.x * 256 + threadIdx.x;
  if (n >= HWHW) return;
  const bf16* p = phiT + (size_t)b * WCH * HWHW;
  const float* w = wk + b * WCH;
  float d = 0.f;
  for (int k = 0; k < WCH; ++k) d += (float)p[(size_t)k * HWHW + n] * w[k];
  invD[(size_t)b * HWHW + n] = rsqrtf(d + 1e-8f);
}

// ---------- K6: t[b][k][c] += sum_n (phiT[k][n]*invD[n]) * X_t[n][c], split-K ----------
__global__ __launch_bounds__(256)
void k_t_gemm(const float* __restrict__ x, const bf16* __restrict__ phiT,
              const float* __restrict__ invD, float* __restrict__ t) {
  const int b = blockIdx.z;
  const int c0 = blockIdx.x * 64;
  const int nbase = blockIdx.y * 224;       // 14 splits * 224 = 3136
  const int tid = threadIdx.x, lane = tid & 31, w = tid >> 5;
  const int wm = w >> 1, wn = w & 1;
  __shared__ __align__(16) bf16 sA[128 * 32];
  __shared__ __align__(16) bf16 sB[64 * 40];
  const float* xb = x + (size_t)b * CHW;
  const bf16* pT = phiT + (size_t)b * WCH * HWHW;
  const float* iD = invD + (size_t)b * HWHW;
  v8f acc[2][2] = {};
  for (int s = 0; s < 7; ++s) {
    int n0 = nbase + s * 32;
    __syncthreads();
#pragma unroll
    for (int i = 0; i < 16; ++i) {           // A: phiT scaled by invD per column
      int e = tid + i * 256, m = e >> 5, nn = e & 31;
      float pv = (float)pT[(size_t)m * HWHW + n0 + nn];
      sA[m * 32 + nn] = (bf16)(pv * iD[n0 + nn]);
    }
#pragma unroll
    for (int i = 0; i < 8; ++i) {            // B: X_t tile, transposed [c][n]
      int e = tid + i * 256, c = e & 63, nn = e >> 6;
      sB[c * 40 + nn] = (bf16)xb[(size_t)(n0 + nn) * CC + c0 + c];
    }
    __syncthreads();
#pragma unroll
    for (int sm = 0; sm < 2; ++sm) {
      v16bf a = frag_A(sA + (wm * 32 + sm * 16) * 32, 32, lane);
#pragma unroll
      for (int sn = 0; sn < 2; ++sn) {
        v16bf bb = frag_B(sB + (wn * 32 + sn * 16) * 40, 40, lane);
        acc[sm][sn] = wmma_bf16(a, bb, acc[sm][sn]);
      }
    }
  }
  float* tb = t + (size_t)b * WCH * CC;
#pragma unroll
  for (int sm = 0; sm < 2; ++sm)
#pragma unroll
    for (int sn = 0; sn < 2; ++sn) {
      int c = c0 + wn * 32 + sn * 16 + (lane & 15);
#pragma unroll
      for (int i = 0; i < 8; ++i) {
        int m = wm * 32 + sm * 16 + i + ((lane < 16) ? 0 : 8);
        atomicAdd(&tb[m * CC + c], acc[sm][sn][i]);
      }
    }
}

// ---------- K7: t3[b][k][co] = (rou[k]*t[k][:]) @ zeta_w^T, bf16 out ----------
__global__ __launch_bounds__(256)
void k_t3_gemm(const float* __restrict__ t, const float* __restrict__ rou,
               const bf16* __restrict__ zw, bf16* __restrict__ t3) {
  const int b = blockIdx.y;
  const int co0 = blockIdx.x * 64;
  const int tid = threadIdx.x, lane = tid & 31, w = tid >> 5;
  const int wm = w >> 1, wn = w & 1;
  __shared__ __align__(16) bf16 sA[128 * 32];
  __shared__ __align__(16) bf16 sB[64 * 40];
  const float* tb = t + (size_t)b * WCH * CC;
  const float* rb = rou + b * WCH;
  v8f acc[2][2] = {};
  for (int k0 = 0; k0 < CC; k0 += 32) {
    __syncthreads();
#pragma unroll
    for (int i = 0; i < 16; ++i) {           // A: rou-scaled t, 128x32
      int e = tid + i * 256, m = e >> 5, c = e & 31;
      sA[m * 32 + c] = (bf16)(tb[m * CC + k0 + c] * rb[m]);
    }
    tile_copy32<64>(zw + (size_t)co0 * CC + k0, CC, sB, 40, tid);  // B: zeta [co][c] (async)
    wait_g2l();
    __syncthreads();
#pragma unroll
    for (int sm = 0; sm < 2; ++sm) {
      v16bf a = frag_A(sA + (wm * 32 + sm * 16) * 32, 32, lane);
#pragma unroll
      for (int sn = 0; sn < 2; ++sn) {
        v16bf bb = frag_B(sB + (wn * 32 + sn * 16) * 40, 40, lane);
        acc[sm][sn] = wmma_bf16(a, bb, acc[sm][sn]);
      }
    }
  }
  bf16* o = t3 + (size_t)b * WCH * CC;
#pragma unroll
  for (int sm = 0; sm < 2; ++sm)
#pragma unroll
    for (int sn = 0; sn < 2; ++sn) {
      int co = co0 + wn * 32 + sn * 16 + (lane & 15);
#pragma unroll
      for (int i = 0; i < 8; ++i) {
        int m = wm * 32 + sm * 16 + i + ((lane < 16) ? 0 : 8);
        o[m * CC + co] = (bf16)acc[sm][sn][i];
      }
    }
}

// ---------- K8: out = x + relu(X_t@zeta^T - invD[n]*(phiT^T@t3) + zb) ----------
// 64(M=n) x 128(N=co) block tile, 2x4 waves of 32x32.
__global__ __launch_bounds__(256)
void k_final(const float* __restrict__ x, const bf16* __restrict__ zw,
             const float* __restrict__ zb, const bf16* __restrict__ phiT,
             const bf16* __restrict__ t3, const float* __restrict__ invD,
             float* __restrict__ out) {
  const int b  = blockIdx.z;
  const int n0 = blockIdx.y * 64;
  const int co0 = blockIdx.x * 128;
  const int tid = threadIdx.x, lane = tid & 31, w = tid >> 5;
  const int wm = w >> 2, wn = w & 3;         // 2x4 waves, 32x32 subtile each
  __shared__ __align__(16) bf16 sA[64 * 40];
  __shared__ __align__(16) bf16 sB[128 * 40];
  const float* xb = x + (size_t)b * CHW;
  const bf16* pT = phiT + (size_t)b * WCH * HWHW;
  const bf16* t3b = t3 + (size_t)b * WCH * CC;
  v8f acc1[2][2] = {}, acc2[2][2] = {};
  // Part 1: K = C = 512 over c (X_t @ zeta^T)
  for (int k0 = 0; k0 < CC; k0 += 32) {
    __syncthreads();
#pragma unroll
    for (int i = 0; i < 8; ++i) {            // A: X_t tile 64x32 (f32 -> bf16)
      int e = tid + i * 256, c = e & 31, n = e >> 5;
      sA[n * 40 + c] = (bf16)xb[(size_t)(n0 + n) * CC + k0 + c];
    }
    tile_copy32<128>(zw + (size_t)co0 * CC + k0, CC, sB, 40, tid);  // B: zeta 128x32 (async)
    wait_g2l();
    __syncthreads();
#pragma unroll
    for (int sm = 0; sm < 2; ++sm) {
      v16bf a = frag_A(sA + (wm * 32 + sm * 16) * 40, 40, lane);
#pragma unroll
      for (int sn = 0; sn < 2; ++sn) {
        v16bf bb = frag_B(sB + (wn * 32 + sn * 16) * 40, 40, lane);
        acc1[sm][sn] = wmma_bf16(a, bb, acc1[sm][sn]);
      }
    }
  }
  // Part 2: K = WCH = 128 over k (phiT^T @ t3)
  for (int k0 = 0; k0 < WCH; k0 += 32) {
    __syncthreads();
#pragma unroll
    for (int i = 0; i < 8; ++i) {            // A2: phiT gather-transpose [n][k]
      int e = tid + i * 256, n = e & 63, k = e >> 6;
      sA[n * 40 + k] = pT[(size_t)(k0 + k) * HWHW + n0 + n];
    }
#pragma unroll
    for (int i = 0; i < 16; ++i) {           // B2: t3 transposed [co][k], 128x32
      int e = tid + i * 256, co = e & 127, k = e >> 7;
      sB[co * 40 + k] = t3b[(size_t)(k0 + k) * CC + co0 + co];
    }
    __syncthreads();
#pragma unroll
    for (int sm = 0; sm < 2; ++sm) {
      v16bf a = frag_A(sA + (wm * 32 + sm * 16) * 40, 40, lane);
#pragma unroll
      for (int sn = 0; sn < 2; ++sn) {
        v16bf bb = frag_B(sB + (wn * 32 + sn * 16) * 40, 40, lane);
        acc2[sm][sn] = wmma_bf16(a, bb, acc2[sm][sn]);
      }
    }
  }
  const float* iD = invD + (size_t)b * HWHW;
#pragma unroll
  for (int sm = 0; sm < 2; ++sm)
#pragma unroll
    for (int sn = 0; sn < 2; ++sn) {
      int co = co0 + wn * 32 + sn * 16 + (lane & 15);
      float zbv = zb[co];
#pragma unroll
      for (int i = 0; i < 8; ++i) {
        int n = n0 + wm * 32 + sm * 16 + i + ((lane < 16) ? 0 : 8);
        float v = fmaxf(acc1[sm][sn][i] + zbv - iD[n] * acc2[sm][sn][i], 0.f);
        size_t idx = (size_t)n * CC + co;
        out[(size_t)b * CHW + idx] = xb[idx] + v;   // residual: same flat index
      }
    }
}

extern "C" void kernel_launch(void* const* d_in, const int* in_sizes, int n_in,
                              void* d_out, int out_size, void* d_ws, size_t ws_size,
                              hipStream_t stream) {
  (void)in_sizes; (void)n_in; (void)out_size; (void)ws_size;
  const float* x      = (const float*)d_in[0];
  const float* y      = (const float*)d_in[1];
  const float* phi_w  = (const float*)d_in[2];
  const float* phi_b  = (const float*)d_in[3];
  const float* rou_w  = (const float*)d_in[4];
  const float* rou_b  = (const float*)d_in[5];
  const float* zeta_w = (const float*)d_in[6];
  const float* zeta_b = (const float*)d_in[7];
  float* out = (float*)d_out;

  char* ws = (char*)d_ws;
  size_t off = 0;
  auto alloc = [&](size_t bytes) -> void* {
    void* p = ws + off;
    off += (bytes + 255) & ~(size_t)255;
    return p;
  };
  bf16*  phiT   = (bf16*)alloc((size_t)BB * WCH * HWHW * 2);  // ~12.25 MB
  float* t      = (float*)alloc((size_t)BB * WCH * CC * 4);   // 4 MB
  bf16*  t3     = (bf16*)alloc((size_t)BB * WCH * CC * 2);    // 2 MB
  bf16*  zwb    = (bf16*)alloc((size_t)CC * CC * 2);
  bf16*  pwb    = (bf16*)alloc((size_t)WCH * CC * 2);
  float* pooled = (float*)alloc((size_t)BB * CC * 4);
  float* rou    = (float*)alloc((size_t)BB * WCH * 4);
  float* sbuf   = (float*)alloc((size_t)BB * WCH * 4);
  float* wk     = (float*)alloc((size_t)BB * WCH * 4);
  float* invD   = (float*)alloc((size_t)BB * HWHW * 4);

  k_cvt_w<<<(CC * CC + 255) / 256, 256, 0, stream>>>(phi_w, zeta_w, pwb, zwb);
  k_pool<<<dim3(CC, BB), 256, 0, stream>>>(y, pooled);
  k_rou<<<BB, WCH, 0, stream>>>(pooled, rou_w, rou_b, rou);
  k_phi_gemm<<<dim3(HWHW / 64, BB), 256, 0, stream>>>(x, pwb, phi_b, phiT);
  k_s<<<dim3(WCH, BB), 256, 0, stream>>>(phiT, rou, sbuf, wk);
  k_invd<<<dim3((HWHW + 255) / 256, BB), 256, 0, stream>>>(phiT, wk, invD);
  (void)hipMemsetAsync(t, 0, (size_t)BB * WCH * CC * 4, stream);
  k_t_gemm<<<dim3(CC / 64, 14, BB), 256, 0, stream>>>(x, phiT, invD, t);
  k_t3_gemm<<<dim3(CC / 64, BB), 256, 0, stream>>>(t, rou, zwb, t3);
  k_final<<<dim3(CC / 128, HWHW / 64, BB), 256, 0, stream>>>(x, zwb, zeta_b, phiT, t3, invD, out);
}
